// Attention_39341900431731
// MI455X (gfx1250) — matrix operand order repaired and checked by reading, not compiled, not run on hardware
//
#include <hip/hip_runtime.h>
#include <hip/hip_bf16.h>
#include <stdint.h>

// ---------------------------------------------------------------------------
// Problem dims (fixed by reference)
// ---------------------------------------------------------------------------
#define BB    256
#define NN    65
#define DIMD  1024
#define HH    16
#define DHD   64
#define MTOT  (BB * NN)          // 16640 = 260 * 64
#define PADK  48                 // padded LDS row stride (elements) for K=32 tiles

typedef __attribute__((ext_vector_type(16))) __bf16 v16bf;
typedef __attribute__((ext_vector_type(8)))  __bf16 v8bf;
typedef __attribute__((ext_vector_type(8)))  float  v8f;

static __device__ __forceinline__ __bf16 f2bf(float x) { return (__bf16)x; }

static __device__ __forceinline__ v8f vzero8() {
  v8f z;
#pragma unroll
  for (int i = 0; i < 8; ++i) z[i] = 0.f;
  return z;
}

// ---------------------------------------------------------------------------
// CDNA5 async global->LDS copy (16B per lane), tracked by ASYNCcnt.
// ---------------------------------------------------------------------------
static __device__ __forceinline__ void async_copy16(const __bf16* g,
                                                    __bf16* l) {
  uint32_t loff =
      (uint32_t)(uintptr_t)(__attribute__((address_space(3))) void*)l;
  asm volatile("global_load_async_to_lds_b128 %0, %1, off"
               :
               : "v"(loff), "v"(g)
               : "memory");
}

static __device__ __forceinline__ void wait_async0() {
  asm volatile("s_wait_asynccnt 0" ::: "memory");
}

// ---------------------------------------------------------------------------
// f32 -> bf16 convert
// ---------------------------------------------------------------------------
__global__ void cvt_f32_bf16(const float* __restrict__ in,
                             __bf16* __restrict__ out, int n) {
  int i = blockIdx.x * blockDim.x + threadIdx.x;
  if (i < n) out[i] = f2bf(in[i]);
}

// ---------------------------------------------------------------------------
// Core bf16 WMMA GEMM with double-buffered async LDS staging.
// Block: 256 threads = 8 waves (2 M x 4 N).  Block tile 64(M) x 256(N).
// A: [M][K=1024] bf16 row-major.  Bw: [N][K=1024] bf16 row-major (C = A*Bw^T).
// ---------------------------------------------------------------------------
struct Tile {
  v8f acc[2][4];
};

static __device__ __forceinline__ void stage_tiles(
    const __bf16* __restrict__ A, const __bf16* __restrict__ Bw, int mblk,
    int nblk, int k0, __bf16* sA, __bf16* sB) {
  int t = threadIdx.x;
  int part = t & 3;                       // 16B segment within a 64B row chunk
  int r = t >> 2;                         // 0..63
  // A tile: 64 rows x 32 cols
  async_copy16(A + (size_t)(mblk + r) * DIMD + k0 + part * 8,
               sA + r * PADK + part * 8);
  // B tile: 256 rows x 32 cols
#pragma unroll
  for (int seg = 0; seg < 4; ++seg) {
    int row = seg * 64 + r;
    async_copy16(Bw + (size_t)(nblk + row) * DIMD + k0 + part * 8,
                 sB + row * PADK + part * 8);
  }
}

static __device__ __forceinline__ Tile gemm_tile_k1024(
    const __bf16* __restrict__ A, const __bf16* __restrict__ Bw, int mblk,
    int nblk, __bf16* sA, __bf16* sB) {
  int lane = threadIdx.x & 31;
  int l = lane & 15;
  int hi = lane >> 4;
  int wave = threadIdx.x >> 5;
  int waveM = wave >> 2, waveN = wave & 3;

  Tile t;
#pragma unroll
  for (int i = 0; i < 2; ++i)
#pragma unroll
    for (int j = 0; j < 4; ++j) t.acc[i][j] = vzero8();

  // preload stage 0
  stage_tiles(A, Bw, mblk, nblk, 0, sA, sB);

  for (int k0 = 0; k0 < DIMD; k0 += 32) {
    int cur = (k0 >> 5) & 1;
    __bf16* cA = sA + cur * (64 * PADK);
    __bf16* cB = sB + cur * (256 * PADK);
    __bf16* nA = sA + (cur ^ 1) * (64 * PADK);
    __bf16* nB = sB + (cur ^ 1) * (256 * PADK);

    wait_async0();        // this wave's staged data landed in LDS
    __syncthreads();      // everyone's landed; prev-stage readers done

    if (k0 + 32 < DIMD) stage_tiles(A, Bw, mblk, nblk, k0 + 32, nA, nB);

    v16bf a[2];
#pragma unroll
    for (int mf = 0; mf < 2; ++mf) {
      const __bf16* p = cA + (waveM * 32 + mf * 16 + l) * PADK + hi * 8;
      v8bf a0 = *(const v8bf*)p;
      v8bf a1 = *(const v8bf*)(p + 16);
#pragma unroll
      for (int i = 0; i < 8; ++i) {
        a[mf][i] = a0[i];
        a[mf][i + 8] = a1[i];
      }
    }
    // One B fragment live at a time: keeps peak VGPR pressure low enough to
    // hold the 2x4 accumulator tile without scratch spills.
#pragma unroll
    for (int nf = 0; nf < 4; ++nf) {
      const __bf16* p = cB + (waveN * 64 + nf * 16 + l) * PADK + hi * 8;
      v8bf b0 = *(const v8bf*)p;
      v8bf b1 = *(const v8bf*)(p + 16);
      v16bf b;
#pragma unroll
      for (int i = 0; i < 8; ++i) {
        b[i] = b0[i];
        b[i + 8] = b1[i];
      }
      t.acc[0][nf] = __builtin_amdgcn_wmma_f32_16x16x32_bf16(
          false, a[0], false, b, (short)0, t.acc[0][nf], false, false);
      t.acc[1][nf] = __builtin_amdgcn_wmma_f32_16x16x32_bf16(
          false, a[1], false, b, (short)0, t.acc[1][nf], false, false);
    }
  }
  return t;
}

// ---------------------------------------------------------------------------
// K1: QKV GEMM.  grid (260, 12), block 256.
// Writes Qb/Kb/Vb bf16 [b,h,n,dh] and Vf f32 [b,n,h,dh].
// ---------------------------------------------------------------------------
__global__ void __launch_bounds__(256, 1)
qkv_gemm(const __bf16* __restrict__ Xb, const __bf16* __restrict__ Wq,
         const float* __restrict__ bias, __bf16* __restrict__ Qb,
         __bf16* __restrict__ Kb, __bf16* __restrict__ Vb,
         float* __restrict__ Vf) {
  __shared__ __align__(32) __bf16 sA[2 * 64 * PADK];
  __shared__ __align__(32) __bf16 sB[2 * 256 * PADK];

  int mblk = blockIdx.x * 64;
  int nblk = blockIdx.y * 256;

  Tile t = gemm_tile_k1024(Xb, Wq, mblk, nblk, sA, sB);

  int lane = threadIdx.x & 31;
  int l = lane & 15;
  int hi = lane >> 4;
  int wave = threadIdx.x >> 5;
  int waveM = wave >> 2, waveN = wave & 3;

#pragma unroll
  for (int mf = 0; mf < 2; ++mf) {
#pragma unroll
    for (int nf = 0; nf < 4; ++nf) {
#pragma unroll
      for (int r = 0; r < 8; ++r) {
        int m = mblk + waveM * 32 + mf * 16 + hi * 8 + r;
        int col = nblk + waveN * 64 + nf * 16 + l;
        float v = t.acc[mf][nf][r] + bias[col];
        int b = m / NN, n = m % NN;
        int which = col >> 10;    // 0=q, 1=k, 2=v
        int within = col & 1023;
        int h = within >> 6, d = within & 63;
        size_t idx = (((size_t)b * HH + h) * NN + n) * DHD + d;
        if (which == 0) {
          Qb[idx] = f2bf(v);
        } else if (which == 1) {
          Kb[idx] = f2bf(v);
        } else {
          Vb[idx] = f2bf(v);
          Vf[(((size_t)b * NN + n) * HH + h) * DHD + d] = v;
        }
      }
    }
  }
}

// ---------------------------------------------------------------------------
// K2: attention per (b,h).  grid 4096, block 160 (5 waves).
// LDS (aliased):
//   [0      , 10240)  sQ  bf16 [80][64]     -> later sP bf16 [80][96]
//   [10240  , 20480)  sK  bf16 [80][64]        (sP spans [0, 15360))
//   [20480  , 32768)  sVT bf16 [64][96]
//   [32768  , 58368)  sD  f32  [80][80]
// ---------------------------------------------------------------------------
__global__ void __launch_bounds__(160, 1)
attn_kernel(const __bf16* __restrict__ Qb, const __bf16* __restrict__ Kb,
            const __bf16* __restrict__ Vb, float* __restrict__ O1) {
  __shared__ __align__(128) unsigned char smem[58368];
  __bf16* sQ = (__bf16*)smem;                   // [80][64]
  __bf16* sK = (__bf16*)(smem + 10240);         // [80][64]
  __bf16* sVT = (__bf16*)(smem + 20480);        // [64][96]
  float* sD = (float*)(smem + 32768);           // [80][80]
  __bf16* sP = (__bf16*)smem;                   // [80][96], aliases sQ/sK

  int bh = blockIdx.x;                          // b*16 + h
  size_t base = (size_t)bh * NN * DHD;

  // ---- stage Q, K (zero-padded rows) and V^T (zero-padded cols) ----
  for (int idx = threadIdx.x; idx < 80 * 64; idx += 160) {
    int row = idx >> 6, d = idx & 63;
    __bf16 zq = (__bf16)0.f, zk = (__bf16)0.f;
    if (row < NN) {
      zq = Qb[base + (size_t)row * DHD + d];
      zk = Kb[base + (size_t)row * DHD + d];
    }
    sQ[idx] = zq;
    sK[idx] = zk;
  }
  for (int idx = threadIdx.x; idx < 64 * 96; idx += 160) {
    int d = idx / 96, j = idx % 96;
    sVT[idx] = (j < NN) ? Vb[base + (size_t)j * DHD + d] : (__bf16)0.f;
  }
  __syncthreads();

  int lane = threadIdx.x & 31;
  int l = lane & 15;
  int hi = lane >> 4;
  int w = threadIdx.x >> 5;                     // wave 0..4 -> M tile

  // ---- dots = Q * K^T  (M=80 pad, N=80 pad, K=64) ----
  v8f accD[5];
#pragma unroll
  for (int nf = 0; nf < 5; ++nf) accD[nf] = vzero8();

#pragma unroll
  for (int k0 = 0; k0 < 64; k0 += 32) {
    v16bf a;
    {
      const __bf16* p = sQ + (w * 16 + l) * 64 + k0 + hi * 8;
      v8bf a0 = *(const v8bf*)p;
      v8bf a1 = *(const v8bf*)(p + 16);
#pragma unroll
      for (int i = 0; i < 8; ++i) {
        a[i] = a0[i];
        a[i + 8] = a1[i];
      }
    }
#pragma unroll
    for (int nf = 0; nf < 5; ++nf) {
      const __bf16* p = sK + (nf * 16 + l) * 64 + k0 + hi * 8;
      v8bf b0 = *(const v8bf*)p;
      v8bf b1 = *(const v8bf*)(p + 16);
      v16bf bfr;
#pragma unroll
      for (int i = 0; i < 8; ++i) {
        bfr[i] = b0[i];
        bfr[i + 8] = b1[i];
      }
      accD[nf] = __builtin_amdgcn_wmma_f32_16x16x32_bf16(
          false, a, false, bfr, (short)0, accD[nf], false, false);
    }
  }

  const float scale = 0.03125f;  // 1024^-0.5
#pragma unroll
  for (int nf = 0; nf < 5; ++nf)
#pragma unroll
    for (int r = 0; r < 8; ++r) {
      int row = w * 16 + hi * 8 + r;
      int col = nf * 16 + l;
      sD[row * 80 + col] = accD[nf][r] * scale;
    }
  __syncthreads();

  // ---- softmax rows (mask j >= 65), write P bf16 with zero K-padding ----
  if (threadIdx.x < 80) {
    int row = threadIdx.x;
    if (row < NN) {
      float m = -1e30f;
      for (int j = 0; j < NN; ++j) m = fmaxf(m, sD[row * 80 + j]);
      float s = 0.f;
      for (int j = 0; j < NN; ++j) s += __expf(sD[row * 80 + j] - m);
      float rs = 1.f / s;
      for (int j = 0; j < 96; ++j)
        sP[row * 96 + j] =
            (j < NN) ? f2bf(__expf(sD[row * 80 + j] - m) * rs) : (__bf16)0.f;
    } else {
      for (int j = 0; j < 96; ++j) sP[row * 96 + j] = (__bf16)0.f;
    }
  }
  __syncthreads();

  // ---- out1 = P * V  (M=80 pad, N=64, K=96 pad) ----
  v8f accO[4];
#pragma unroll
  for (int nf = 0; nf < 4; ++nf) accO[nf] = vzero8();

#pragma unroll
  for (int k0 = 0; k0 < 96; k0 += 32) {
    v16bf a;
    {
      const __bf16* p = sP + (w * 16 + l) * 96 + k0 + hi * 8;
      v8bf a0 = *(const v8bf*)p;
      v8bf a1 = *(const v8bf*)(p + 16);
#pragma unroll
      for (int i = 0; i < 8; ++i) {
        a[i] = a0[i];
        a[i + 8] = a1[i];
      }
    }
#pragma unroll
    for (int nf = 0; nf < 4; ++nf) {
      const __bf16* p = sVT + (nf * 16 + l) * 96 + k0 + hi * 8;
      v8bf b0 = *(const v8bf*)p;
      v8bf b1 = *(const v8bf*)(p + 16);
      v16bf bfr;
#pragma unroll
      for (int i = 0; i < 8; ++i) {
        bfr[i] = b0[i];
        bfr[i + 8] = b1[i];
      }
      accO[nf] = __builtin_amdgcn_wmma_f32_16x16x32_bf16(
          false, a, false, bfr, (short)0, accO[nf], false, false);
    }
  }

#pragma unroll
  for (int nf = 0; nf < 4; ++nf)
#pragma unroll
    for (int r = 0; r < 8; ++r) {
      int row = w * 16 + hi * 8 + r;
      if (row < NN)
        O1[base + (size_t)row * DHD + nf * 16 + l] = accO[nf][r];
    }
}

// ---------------------------------------------------------------------------
// K3: conv 3x3 (SAME) over channels=65, image 16x64, + BN + residual add.
// grid (65, 256) = (out_channel, batch), block 256. Writes A2 bf16 [b,n,(h d)].
// ---------------------------------------------------------------------------
__global__ void conv_bn_kernel(const float* __restrict__ Vf,
                               const float* __restrict__ cw,
                               const float* __restrict__ cb,
                               const float* __restrict__ gamma,
                               const float* __restrict__ beta,
                               const float* __restrict__ mean,
                               const float* __restrict__ var,
                               const float* __restrict__ O1,
                               __bf16* __restrict__ A2) {
  int co = blockIdx.x;   // out channel 0..64
  int b = blockIdx.y;    // batch

  __shared__ float sw[NN * 9];
  for (int i = threadIdx.x; i < NN * 9; i += blockDim.x)
    sw[i] = cw[(size_t)co * NN * 9 + i];
  __syncthreads();

  float inv = rsqrtf(var[co] + 1e-5f) * gamma[co];
  float cbias = cb[co];
  float mu = mean[co];
  float bt = beta[co];

  const float* vbase = Vf + (size_t)b * NN * (HH * DHD);

  for (int p = threadIdx.x; p < HH * DHD; p += blockDim.x) {
    int h = p >> 6, x = p & 63;
    float acc = 0.f;
    for (int ci = 0; ci < NN; ++ci) {
      const float* vin = vbase + (size_t)ci * (HH * DHD);
      const float* wr = sw + ci * 9;
#pragma unroll
      for (int kh = 0; kh < 3; ++kh) {
        int hh = h + kh - 1;
        if ((unsigned)hh >= (unsigned)HH) continue;
#pragma unroll
        for (int kw = 0; kw < 3; ++kw) {
          int xx = x + kw - 1;
          if ((unsigned)xx >= (unsigned)DHD) continue;
          acc = fmaf(vin[hh * DHD + xx], wr[kh * 3 + kw], acc);
        }
      }
    }
    float bn = (acc + cbias - mu) * inv + bt;
    float o1 = O1[(((size_t)b * HH + h) * NN + co) * DHD + x];
    A2[((size_t)b * NN + co) * DIMD + h * DHD + x] = f2bf(bn + o1);
  }
}

// ---------------------------------------------------------------------------
// K4: output GEMM.  grid (260, 4), block 256.  Y[m][col] f32.
// ---------------------------------------------------------------------------
__global__ void __launch_bounds__(256, 1)
out_gemm(const __bf16* __restrict__ A2, const __bf16* __restrict__ Wo,
         const float* __restrict__ bias, float* __restrict__ Y) {
  __shared__ __align__(32) __bf16 sA[2 * 64 * PADK];
  __shared__ __align__(32) __bf16 sB[2 * 256 * PADK];

  int mblk = blockIdx.x * 64;
  int nblk = blockIdx.y * 256;

  Tile t = gemm_tile_k1024(A2, Wo, mblk, nblk, sA, sB);

  int lane = threadIdx.x & 31;
  int l = lane & 15;
  int hi = lane >> 4;
  int wave = threadIdx.x >> 5;
  int waveM = wave >> 2, waveN = wave & 3;

#pragma unroll
  for (int mf = 0; mf < 2; ++mf)
#pragma unroll
    for (int nf = 0; nf < 4; ++nf)
#pragma unroll
      for (int r = 0; r < 8; ++r) {
        int m = mblk + waveM * 32 + mf * 16 + hi * 8 + r;
        int col = nblk + waveN * 64 + nf * 16 + l;
        Y[(size_t)m * DIMD + col] = t.acc[mf][nf][r] + bias[col];
      }
}

// ---------------------------------------------------------------------------
// Host launcher
// ---------------------------------------------------------------------------
extern "C" void kernel_launch(void* const* d_in, const int* in_sizes, int n_in,
                              void* d_out, int out_size, void* d_ws,
                              size_t ws_size, hipStream_t stream) {
  const float* x = (const float*)d_in[0];
  const float* w_qkv = (const float*)d_in[1];
  const float* b_qkv = (const float*)d_in[2];
  const float* w_out = (const float*)d_in[3];
  const float* b_out = (const float*)d_in[4];
  const float* conv_w = (const float*)d_in[5];
  const float* conv_b = (const float*)d_in[6];
  const float* bn_g = (const float*)d_in[7];
  const float* bn_b = (const float*)d_in[8];
  const float* bn_m = (const float*)d_in[9];
  const float* bn_v = (const float*)d_in[10];

  char* ws = (char*)d_ws;
  size_t off = 0;
  auto alloc = [&](size_t bytes) -> char* {
    char* p = ws + off;
    off = (off + bytes + 255) & ~(size_t)255;
    return p;
  };

  __bf16* xb = (__bf16*)alloc((size_t)MTOT * DIMD * 2);
  __bf16* wqb = (__bf16*)alloc((size_t)3 * DIMD * DIMD * 2);
  __bf16* wob = (__bf16*)alloc((size_t)DIMD * DIMD * 2);
  __bf16* Qb = (__bf16*)alloc((size_t)BB * HH * NN * DHD * 2);
  __bf16* Kb = (__bf16*)alloc((size_t)BB * HH * NN * DHD * 2);
  __bf16* Vb = (__bf16*)alloc((size_t)BB * HH * NN * DHD * 2);
  float* Vf = (float*)alloc((size_t)BB * NN * HH * DHD * 4);
  float* O1 = (float*)alloc((size_t)BB * HH * NN * DHD * 4);
  __bf16* A2 = (__bf16*)alloc((size_t)MTOT * DIMD * 2);

  // converts
  {
    int n = MTOT * DIMD;
    cvt_f32_bf16<<<(n + 255) / 256, 256, 0, stream>>>(x, xb, n);
  }
  {
    int n = 3 * DIMD * DIMD;
    cvt_f32_bf16<<<(n + 255) / 256, 256, 0, stream>>>(w_qkv, wqb, n);
  }
  {
    int n = DIMD * DIMD;
    cvt_f32_bf16<<<(n + 255) / 256, 256, 0, stream>>>(w_out, wob, n);
  }

  qkv_gemm<<<dim3(MTOT / 64, (3 * DIMD) / 256), 256, 0, stream>>>(
      xb, wqb, b_qkv, Qb, Kb, Vb, Vf);

  attn_kernel<<<BB * HH, 160, 0, stream>>>(Qb, Kb, Vb, O1);

  conv_bn_kernel<<<dim3(NN, BB), 256, 0, stream>>>(Vf, conv_w, conv_b, bn_g,
                                                   bn_b, bn_m, bn_v, O1, A2);

  out_gemm<<<dim3(MTOT / 64, DIMD / 256), 256, 0, stream>>>(A2, wob, b_out,
                                                            (float*)d_out);
}
